// Model_16441134809434
// MI455X (gfx1250) — compile-verified, matrix operand order checked by reading, not run on hardware
//
#include <hip/hip_runtime.h>
#include <hip/hip_bf16.h>

typedef __bf16 bf16;
typedef __attribute__((ext_vector_type(16))) __bf16 v16bf;
typedef __attribute__((ext_vector_type(8)))  float  v8f;
typedef __attribute__((ext_vector_type(4)))  int    v4i;

#define DD    768
#define HH    8
#define DQ    96
#define LL    512
#define NLAY  6
#define FFD   3072
#define NREL  199
#define NB    8
#define TT    4096
#define NOUT  1024

// ---------------------------------------------------------------------------
// Generic batched GEMM: C[z] (fp32, MxN, ldc) = A[z] (bf16 MxK, lda) x B[z] (bf16 KxN, ldb)
// Block tile 64x128, 256 threads = 8 waves (2x4), each wave owns a 32x32 output
// tile: 2 A-frags x 2 B-frags -> 4 x v_wmma_f32_16x16x32_bf16 per 32-K chunk.
// A tile is staged with GLOBAL_LOAD_ASYNC_TO_LDS_B128 (ASYNCcnt) when the
// toolchain exposes the builtin; B tile is transposed into LDS manually.
// Requirements: M % 64 == 0, K % 32 == 0, lda % 8 == 0, A base 16B aligned.
// N arbitrary (guarded). accum=1 -> C += A*B. Prefetches next K-chunk tiles.
// ---------------------------------------------------------------------------
__global__ __launch_bounds__(256) void k_gemm(
    const bf16* __restrict__ A, const bf16* __restrict__ Bm, float* __restrict__ C,
    int M, int N, int K, int lda, int ldb, int ldc,
    long sA, long sB, long sC, int zmodB, int accum)
{
  int z = blockIdx.z;
  A  += (long)z * sA;
  Bm += (zmodB ? (long)(z % zmodB) : (long)z) * sB;
  C  += (long)z * sC;

  __shared__ __align__(16) bf16 lA[64][32];
  __shared__ __align__(16) bf16 lB[128][32];   // transposed: lB[n][k]

  int tid  = threadIdx.x;
  int wave = tid >> 5, lane = tid & 31;
  int mw = wave >> 2, nw = wave & 3;           // 2 x 4 wave grid
  int half = lane >> 4, l16 = lane & 15;
  int row0 = blockIdx.y * 64, col0 = blockIdx.x * 128;

  // global staging coordinates
  int ar = tid >> 2, ac = (tid & 3) * 8;       // A: 64 x 32
  const bf16* gA = A + (long)(row0 + ar) * lda + ac;

  v8f acc00 = {}, acc01 = {}, acc10 = {}, acc11 = {};
  if (accum) {
    for (int r = 0; r < 8; ++r) {
      int m0 = row0 + mw * 32 + half * 8 + r;
      int n0 = col0 + nw * 32 + l16;
      if (n0 < N)      acc00[r] = C[(long)m0 * ldc + n0];
      if (n0 + 16 < N) acc01[r] = C[(long)m0 * ldc + n0 + 16];
      if (n0 < N)      acc10[r] = C[(long)(m0 + 16) * ldc + n0];
      if (n0 + 16 < N) acc11[r] = C[(long)(m0 + 16) * ldc + n0 + 16];
    }
  }

  for (int kk = 0; kk < K; kk += 32) {
    // ---- stage A tile 64x32: async global->LDS when available ----
#if __has_builtin(__builtin_amdgcn_global_load_async_to_lds_b128)
    __builtin_amdgcn_global_load_async_to_lds_b128(
        (__attribute__((address_space(1))) v4i*)(gA + kk),
        (__attribute__((address_space(3))) v4i*)&lA[ar][ac],
        0, 0);
#else
    *(uint4*)&lA[ar][ac] = *(const uint4*)(gA + kk);
#endif
    // ---- stage B tile 32x128 transposed (two 16B chunks per thread) ----
    for (int j = 0; j < 2; ++j) {
      int lin = tid + j * 256;
      int br = lin >> 4, bc = (lin & 15) * 8;
      union { uint4 q; bf16 e[8]; } t;
      const bf16* src = Bm + (long)(kk + br) * ldb + col0 + bc;
      if (((ldb & 7) == 0) && (col0 + bc + 8 <= N)) {
        t.q = *(const uint4*)src;
      } else {
        for (int i = 0; i < 8; ++i) t.e[i] = (col0 + bc + i < N) ? src[i] : (bf16)0.f;
      }
      for (int i = 0; i < 8; ++i) lB[bc + i][br] = t.e[i];
    }
    // ---- prefetch next K-chunk into caches (global_prefetch_b8) ----
    if (kk + 32 < K) {
      __builtin_prefetch(gA + kk + 32, 0, 3);
      __builtin_prefetch(Bm + (long)(kk + 32 + (tid >> 4)) * ldb + col0 + (tid & 15) * 8, 0, 3);
    }
#if __has_builtin(__builtin_amdgcn_global_load_async_to_lds_b128)
#if __has_builtin(__builtin_amdgcn_s_wait_asynccnt)
    __builtin_amdgcn_s_wait_asynccnt(0);
#else
    asm volatile("s_wait_asynccnt 0" ::: "memory");
#endif
#endif
    __syncthreads();

    union { v16bf v; uint4 q[2]; } fa0, fa1, fb0, fb1;
    int m = mw * 32 + l16;
    fa0.q[0] = *(uint4*)&lA[m][half * 8];
    fa0.q[1] = *(uint4*)&lA[m][16 + half * 8];
    fa1.q[0] = *(uint4*)&lA[m + 16][half * 8];
    fa1.q[1] = *(uint4*)&lA[m + 16][16 + half * 8];
    int n0 = nw * 32 + l16;
    fb0.q[0] = *(uint4*)&lB[n0][half * 8];
    fb0.q[1] = *(uint4*)&lB[n0][16 + half * 8];
    fb1.q[0] = *(uint4*)&lB[n0 + 16][half * 8];
    fb1.q[1] = *(uint4*)&lB[n0 + 16][16 + half * 8];

    acc00 = __builtin_amdgcn_wmma_f32_16x16x32_bf16(false, fa0.v, false, fb0.v,
                                                    (short)0, acc00, false, false);
    acc01 = __builtin_amdgcn_wmma_f32_16x16x32_bf16(false, fa0.v, false, fb1.v,
                                                    (short)0, acc01, false, false);
    acc10 = __builtin_amdgcn_wmma_f32_16x16x32_bf16(false, fa1.v, false, fb0.v,
                                                    (short)0, acc10, false, false);
    acc11 = __builtin_amdgcn_wmma_f32_16x16x32_bf16(false, fa1.v, false, fb1.v,
                                                    (short)0, acc11, false, false);
    __syncthreads();
  }

  for (int r = 0; r < 8; ++r) {
    int m0 = row0 + mw * 32 + half * 8 + r;
    int n0 = col0 + nw * 32 + l16;
    if (n0 < N)      C[(long)m0 * ldc + n0]             = acc00[r];
    if (n0 + 16 < N) C[(long)m0 * ldc + n0 + 16]        = acc01[r];
    if (n0 < N)      C[(long)(m0 + 16) * ldc + n0]      = acc10[r];
    if (n0 + 16 < N) C[(long)(m0 + 16) * ldc + n0 + 16] = acc11[r];
  }
}

// ---------------------------------------------------------------------------
// Generic strided fp32 -> bf16 pack / transpose / head-shuffle.
// offset = (z/zdiv)*Outer + (z%zdiv)*Inner for both sides.
// ---------------------------------------------------------------------------
__global__ __launch_bounds__(256) void k_pack(
    const float* __restrict__ in, bf16* __restrict__ out,
    int R, int C, int zdiv,
    long iO, long iI, long iRS, long iCS,
    long oO, long oI, long oRS, long oCS)
{
  long idx = (long)blockIdx.x * 256 + threadIdx.x;
  long tot = (long)R * C;
  if (idx >= tot) return;
  int z = blockIdx.y;
  long r = idx / C, c = idx % C;
  long zi = (long)(z / zdiv) * iO + (long)(z % zdiv) * iI;
  long zo = (long)(z / zdiv) * oO + (long)(z % zdiv) * oI;
  out[zo + r * oRS + c * oCS] = (bf16)in[zi + r * iRS + c * iCS];
}

__global__ __launch_bounds__(256) void k_zero16(bf16* p, long n) {
  long i = (long)blockIdx.x * 256 + threadIdx.x;
  if (i < n) p[i] = (bf16)0.f;
}

// BN(+conv bias) epilogue; optional second (residual) branch; ReLU; write bf16
// into padded time-major buffer (skips pad row 0).
__global__ __launch_bounds__(256) void k_bnep(
    const float* __restrict__ yc, const float* __restrict__ cb, const float* __restrict__ bn,
    const float* __restrict__ yr, const float* __restrict__ cbr, const float* __restrict__ bnr,
    bf16* __restrict__ out, int M, long ycBS, long outBS)
{
  long idx = (long)blockIdx.x * 256 + threadIdx.x;
  long tot = (long)M * DD;
  if (idx >= tot) return;
  int z = blockIdx.y;
  int c = (int)(idx % DD);
  float v = yc[(long)z * ycBS + idx] + cb[c];
  v = bn[c] * (v - bn[2 * DD + c]) * rsqrtf(bn[3 * DD + c] + 1e-5f) + bn[DD + c];
  if (yr) {
    float w = yr[(long)z * ycBS + idx] + cbr[c];
    v += bnr[c] * (w - bnr[2 * DD + c]) * rsqrtf(bnr[3 * DD + c] + 1e-5f) + bnr[DD + c];
  }
  v = fmaxf(v, 0.f);
  out[(long)z * outBS + DD + idx] = (bf16)v;
}

// y += bias (per column), optional relu, optional bf16 mirror
__global__ __launch_bounds__(256) void k_bias_act(
    float* __restrict__ y, const float* __restrict__ bias,
    long total, int N, int relu, bf16* __restrict__ ob)
{
  long idx = (long)blockIdx.x * 256 + threadIdx.x;
  if (idx >= total) return;
  int c = (int)(idx % N);
  float v = y[idx] + bias[c];
  if (relu) v = fmaxf(v, 0.f);
  y[idx] = v;
  if (ob) ob[idx] = (bf16)v;
}

// combine content+relative logits, mask, softmax over j, write bf16 probs
__global__ __launch_bounds__(256) void k_softmax_rel(
    const float* __restrict__ S, const float* __restrict__ R,
    bf16* __restrict__ P, float inv)
{
  int i = blockIdx.x;                 // query position 0..511
  long z = blockIdx.y;                // b*H + h
  int tid = threadIdx.x;
  const float* s = S + (z * LL + i) * LL;
  const float* r = R + (z * LL + i) * NREL;
  __shared__ float red[256];
  float v[2];
  for (int t = 0; t < 2; ++t) {
    int j = tid + t * 256;
    int d = j - i + 99;
    float p = (d >= 0 && d < NREL) ? r[d] : -1e8f;
    v[t] = s[j] * inv + p;
  }
  float m = fmaxf(v[0], v[1]);
  red[tid] = m; __syncthreads();
  for (int o = 128; o > 0; o >>= 1) { if (tid < o) red[tid] = fmaxf(red[tid], red[tid + o]); __syncthreads(); }
  m = red[0]; __syncthreads();
  float e0 = __expf(v[0] - m), e1 = __expf(v[1] - m);
  red[tid] = e0 + e1; __syncthreads();
  for (int o = 128; o > 0; o >>= 1) { if (tid < o) red[tid] += red[tid + o]; __syncthreads(); }
  float rs = 1.f / red[0];
  bf16* p = P + (z * LL + i) * LL;
  p[tid]       = (bf16)(e0 * rs);
  p[tid + 256] = (bf16)(e1 * rs);
}

// x = LayerNorm(x + h); writes fp32 (in place) and bf16 mirror
__global__ __launch_bounds__(256) void k_ln(
    float* __restrict__ x, const float* __restrict__ h,
    const float* __restrict__ g, const float* __restrict__ b, bf16* __restrict__ xb)
{
  int row = blockIdx.x, tid = threadIdx.x;
  long base = (long)row * DD;
  __shared__ float red[256];
  float v[3];
  for (int t = 0; t < 3; ++t) v[t] = x[base + tid + t * 256] + h[base + tid + t * 256];
  red[tid] = v[0] + v[1] + v[2]; __syncthreads();
  for (int o = 128; o > 0; o >>= 1) { if (tid < o) red[tid] += red[tid + o]; __syncthreads(); }
  float mu = red[0] / (float)DD; __syncthreads();
  float q = 0.f;
  for (int t = 0; t < 3; ++t) { v[t] -= mu; q += v[t] * v[t]; }
  red[tid] = q; __syncthreads();
  for (int o = 128; o > 0; o >>= 1) { if (tid < o) red[tid] += red[tid + o]; __syncthreads(); }
  float rs = rsqrtf(red[0] / (float)DD + 1e-5f);
  for (int t = 0; t < 3; ++t) {
    int c = tid + t * 256;
    float o = g[c] * v[t] * rs + b[c];
    x[base + c] = o;
    xb[base + c] = (bf16)o;
  }
}

// ---------------------------------------------------------------------------
extern "C" void kernel_launch(void* const* d_in, const int* in_sizes, int n_in,
                              void* d_out, int out_size, void* d_ws, size_t ws_size,
                              hipStream_t stream)
{
  (void)in_sizes; (void)n_in; (void)out_size; (void)ws_size;
  const float* x_raw    = (const float*)d_in[0];
  const float* rb1_c1w  = (const float*)d_in[1];
  const float* rb1_c1b  = (const float*)d_in[2];
  const float* rb1_bn1  = (const float*)d_in[3];
  const float* rb1_c2w  = (const float*)d_in[4];
  const float* rb1_c2b  = (const float*)d_in[5];
  const float* rb1_bn2  = (const float*)d_in[6];
  const float* rb1_rw   = (const float*)d_in[7];
  const float* rb1_rb   = (const float*)d_in[8];
  const float* rb1_bnr  = (const float*)d_in[9];
  const float* rb23_c1w = (const float*)d_in[10];
  const float* rb23_c1b = (const float*)d_in[11];
  const float* rb23_bn1 = (const float*)d_in[12];
  const float* rb23_c2w = (const float*)d_in[13];
  const float* rb23_c2b = (const float*)d_in[14];
  const float* rb23_bn2 = (const float*)d_in[15];
  const float* rb23_rw  = (const float*)d_in[16];
  const float* rb23_rb  = (const float*)d_in[17];
  const float* rb23_bnr = (const float*)d_in[18];
  const float* w_in_w   = (const float*)d_in[19];
  const float* w_in_b   = (const float*)d_in[20];
  const float* wq       = (const float*)d_in[21];
  const float* wk       = (const float*)d_in[22];
  const float* wv       = (const float*)d_in[23];
  const float* wo       = (const float*)d_in[24];
  const float* rel_emb  = (const float*)d_in[25];
  const float* ln1_g    = (const float*)d_in[26];
  const float* ln1_b    = (const float*)d_in[27];
  const float* ln2_g    = (const float*)d_in[28];
  const float* ln2_b    = (const float*)d_in[29];
  const float* ff1_w    = (const float*)d_in[30];
  const float* ff1_b    = (const float*)d_in[31];
  const float* ff2_w    = (const float*)d_in[32];
  const float* ff2_b    = (const float*)d_in[33];
  const float* w_out_w  = (const float*)d_in[34];
  const float* w_out_b  = (const float*)d_in[35];
  float* out = (float*)d_out;

  // ---- workspace carve ----
  char* ws = (char*)d_ws; size_t off = 0;
  auto alloc = [&](size_t bytes) -> void* {
    off = (off + 255) & ~(size_t)255; void* p = ws + off; off += bytes; return p;
  };
  const long padBS = 2050L * DD;                      // padded time-major batch stride
  bf16* xtm0 = (bf16*)alloc((size_t)NB * 4098 * 32 * 2);
  bf16* bufA = (bf16*)alloc((size_t)NB * padBS * 2);
  bf16* bufB = (bf16*)alloc((size_t)NB * padBS * 2);
  bf16* htm  = (bf16*)alloc((size_t)NB * padBS * 2);
  float* Yc  = (float*)alloc((size_t)NB * 2048 * DD * 4);
  float* Yr  = (float*)alloc((size_t)NB * 2048 * DD * 4);
  bf16* wslot0 = (bf16*)alloc((size_t)3 * 768 * DD * 2);  // conv k-taps (padded-CI fits)
  bf16* wslot1 = (bf16*)alloc((size_t)3 * DD * DD * 2);
  bf16* wslot2 = (bf16*)alloc((size_t)DD * DD * 2);
  bf16* wlin   = (bf16*)alloc((size_t)DD * DD * 2);
  float* xseq  = (float*)alloc((size_t)NB * LL * DD * 4);
  bf16* xbf    = (bf16*)alloc((size_t)NB * LL * DD * 2);
  bf16* wqt    = (bf16*)alloc((size_t)DD * DD * 2);
  bf16* wkt    = (bf16*)alloc((size_t)DD * DD * 2);
  bf16* wvt    = (bf16*)alloc((size_t)DD * DD * 2);
  bf16* wot    = (bf16*)alloc((size_t)DD * DD * 2);
  bf16* relt   = (bf16*)alloc((size_t)HH * DQ * NREL * 2);
  bf16* wff1   = (bf16*)alloc((size_t)DD * FFD * 2);
  bf16* wff2   = (bf16*)alloc((size_t)FFD * DD * 2);
  bf16* woutT  = (bf16*)alloc((size_t)DD * NOUT * 2);
  float* qf    = (float*)alloc((size_t)NB * LL * DD * 4);
  float* kf    = (float*)alloc((size_t)NB * LL * DD * 4);
  float* vf    = (float*)alloc((size_t)NB * LL * DD * 4);
  bf16* qbf    = (bf16*)alloc((size_t)NB * HH * LL * DQ * 2);
  bf16* kbfT   = (bf16*)alloc((size_t)NB * HH * DQ * LL * 2);
  bf16* vbf    = (bf16*)alloc((size_t)NB * HH * LL * DQ * 2);
  float* logit = (float*)alloc((size_t)NB * HH * LL * LL * 4);
  float* rell  = (float*)alloc((size_t)NB * HH * LL * NREL * 4);
  bf16* probs  = (bf16*)alloc((size_t)NB * HH * LL * LL * 2);
  float* attno = (float*)alloc((size_t)NB * HH * LL * DQ * 4);
  bf16* obf    = (bf16*)alloc((size_t)NB * LL * DD * 2);
  float* attnO = (float*)alloc((size_t)NB * LL * DD * 4);
  float* ffh   = (float*)alloc((size_t)NB * LL * FFD * 4);
  bf16* ffhbf  = (bf16*)alloc((size_t)NB * LL * FFD * 2);
  float* ffo   = (float*)alloc((size_t)NB * LL * DD * 4);

  // ---- launch helpers ----
  auto gemm = [&](const bf16* A, const bf16* Bm, float* C, int M, int N, int K,
                  int lda, int ldb, int ldc, int Z, long sA, long sB, long sC,
                  int zmodB, int accum) {
    dim3 g((N + 127) / 128, M / 64, Z);
    k_gemm<<<g, 256, 0, stream>>>(A, Bm, C, M, N, K, lda, ldb, ldc, sA, sB, sC, zmodB, accum);
  };
  auto pack = [&](const float* in, bf16* o, int R, int C, int Z, int zdiv,
                  long iO, long iI, long iRS, long iCS,
                  long oO, long oI, long oRS, long oCS) {
    long tot = (long)R * C;
    dim3 g((unsigned)((tot + 255) / 256), Z);
    k_pack<<<g, 256, 0, stream>>>(in, o, R, C, zdiv, iO, iI, iRS, iCS, oO, oI, oRS, oCS);
  };
  auto zero16 = [&](bf16* p, long n) {
    k_zero16<<<(unsigned)((n + 255) / 256), 256, 0, stream>>>(p, n);
  };

  // ---- stage 0: input to padded time-major bf16 (8 real ch zero-padded to 32) ----
  zero16(xtm0, (long)NB * 4098 * 32);
  zero16(bufA, (long)NB * padBS);
  zero16(bufB, (long)NB * padBS);
  pack(x_raw, xtm0 + 32, TT, 8, NB, 1,
       (long)TT * 8, 0, 8, 1,
       4098L * 32, 0, 32, 1);

  // ---- resblock as GEMMs ----
  auto resblock = [&](const bf16* inb, int ldin, int CI, int CIp, long inBS, int Min,
                      const float* c1w, const float* c1b, const float* bn1,
                      const float* c2w, const float* c2b, const float* bn2,
                      const float* rw, const float* rbias, const float* bnr,
                      bf16* outb) {
    int Mout = Min / 2;
    long ycBS = (long)Mout * DD;
    // conv1 weights -> [3][CIp][DD]
    if (CIp != CI) zero16(wslot0, 3L * CIp * DD);
    pack(c1w, wslot0, CI, DD, 3, 1, 1, 0, 3, (long)CI * 3, (long)CIp * DD, 0, DD, 1);
    for (int k = 0; k < 3; ++k)
      gemm(inb + (long)k * ldin, wslot0 + (long)k * CIp * DD, Yc,
           Mout, DD, CIp, 2 * ldin, DD, DD, NB, inBS, 0, ycBS, 0, k > 0);
    zero16(htm, (long)NB * padBS);
    {
      dim3 g((unsigned)((ycBS + 255) / 256), NB);
      k_bnep<<<g, 256, 0, stream>>>(Yc, c1b, bn1, nullptr, nullptr, nullptr,
                                    htm, Mout, ycBS, padBS);
    }
    // conv2 weights -> [3][DD][DD]
    pack(c2w, wslot1, DD, DD, 3, 1, 1, 0, 3, 3L * DD, (long)DD * DD, 0, DD, 1);
    for (int k = 0; k < 3; ++k)
      gemm(htm + (long)k * DD, wslot1 + (long)k * DD * DD, Yc,
           Mout, DD, DD, DD, DD, DD, NB, padBS, 0, ycBS, 0, k > 0);
    // shortcut 1x1 stride2
    if (CIp != CI) zero16(wslot2, (long)CIp * DD);
    pack(rw, wslot2, CI, DD, 1, 1, 0, 0, 1, CI, 0, 0, DD, 1);
    gemm(inb + ldin, wslot2, Yr, Mout, DD, CIp, 2 * ldin, DD, DD, NB, inBS, 0, ycBS, 0, 0);
    {
      dim3 g((unsigned)((ycBS + 255) / 256), NB);
      k_bnep<<<g, 256, 0, stream>>>(Yc, c2b, bn2, Yr, rbias, bnr, outb, Mout, ycBS, padBS);
    }
  };

  resblock(xtm0, 32, 8, 32, 4098L * 32, TT,
           rb1_c1w, rb1_c1b, rb1_bn1, rb1_c2w, rb1_c2b, rb1_bn2,
           rb1_rw, rb1_rb, rb1_bnr, bufA);
  for (int i = 0; i < 2; ++i) {
    const bf16* inb = (i == 0) ? bufA : bufB;
    bf16* ob = (i == 0) ? bufB : bufA;
    int Min = (i == 0) ? 2048 : 1024;
    resblock(inb, DD, DD, DD, padBS, Min,
             rb23_c1w + (long)i * DD * DD * 3, rb23_c1b + (long)i * DD, rb23_bn1 + (long)i * 4 * DD,
             rb23_c2w + (long)i * DD * DD * 3, rb23_c2b + (long)i * DD, rb23_bn2 + (long)i * 4 * DD,
             rb23_rw + (long)i * DD * DD, rb23_rb + (long)i * DD, rb23_bnr + (long)i * 4 * DD,
             ob);
  }
  // bufA holds [B][2050][DD] (rows 1..512 valid), relu'd

  // ---- input projection: x @ w_in^T + b ----
  pack(w_in_w, wlin, DD, DD, 1, 1, 0, 0, DD, 1, 0, 0, 1, DD);   // transpose
  gemm(bufA + DD, wlin, xseq, LL, DD, DD, DD, DD, DD, NB, padBS, 0, (long)LL * DD, 0, 0);
  {
    long tot = (long)NB * LL * DD;
    k_bias_act<<<(unsigned)((tot + 255) / 256), 256, 0, stream>>>(xseq, w_in_b, tot, DD, 0, xbf);
  }

  // ---- transformer layers ----
  const long M4 = (long)NB * LL;             // 4096
  const float inv_sqrt = 0.1020620726159658f; // 1/sqrt(96)
  for (int l = 0; l < NLAY; ++l) {
    const float* wql = wq + (long)l * HH * DD * DQ;
    const float* wkl = wk + (long)l * HH * DD * DQ;
    const float* wvl = wv + (long)l * HH * DD * DQ;
    const float* wol = wo + (long)l * HH * DQ * DD;
    const float* rll = rel_emb + (long)l * HH * NREL * DQ;

    // weight packs
    pack(wql, wqt, DD, DQ, HH, 1, (long)DD * DQ, 0, DQ, 1, DQ, 0, DD, 1);
    pack(wkl, wkt, DD, DQ, HH, 1, (long)DD * DQ, 0, DQ, 1, DQ, 0, DD, 1);
    pack(wvl, wvt, DD, DQ, HH, 1, (long)DD * DQ, 0, DQ, 1, DQ, 0, DD, 1);
    pack(wol, wot, HH * DQ, DD, 1, 1, 0, 0, DD, 1, 0, 0, DD, 1);
    pack(rll, relt, NREL, DQ, HH, 1, (long)NREL * DQ, 0, DQ, 1, (long)DQ * NREL, 0, 1, NREL);
    pack(ff1_w + (long)l * FFD * DD, wff1, FFD, DD, 1, 1, 0, 0, DD, 1, 0, 0, 1, FFD);
    pack(ff2_w + (long)l * DD * FFD, wff2, DD, FFD, 1, 1, 0, 0, FFD, 1, 0, 0, 1, DD);

    // Q, K, V projections
    gemm(xbf, wqt, qf, (int)M4, DD, DD, DD, DD, DD, 1, 0, 0, 0, 0, 0);
    gemm(xbf, wkt, kf, (int)M4, DD, DD, DD, DD, DD, 1, 0, 0, 0, 0, 0);
    gemm(xbf, wvt, vf, (int)M4, DD, DD, DD, DD, DD, 1, 0, 0, 0, 0, 0);

    // head-layout shuffles to bf16
    pack(qf, qbf, LL, DQ, NB * HH, HH,
         (long)LL * DD, DQ, DD, 1,
         (long)HH * LL * DQ, (long)LL * DQ, DQ, 1);
    pack(kf, kbfT, LL, DQ, NB * HH, HH,
         (long)LL * DD, DQ, DD, 1,
         (long)HH * DQ * LL, (long)DQ * LL, 1, LL);
    pack(vf, vbf, LL, DQ, NB * HH, HH,
         (long)LL * DD, DQ, DD, 1,
         (long)HH * LL * DQ, (long)LL * DQ, DQ, 1);

    // content logits: Q x K^T  [z = b*H+h]
    gemm(qbf, kbfT, logit, LL, LL, DQ, DQ, LL, LL,
         NB * HH, (long)LL * DQ, (long)DQ * LL, (long)LL * LL, 0, 0);
    // relative logits: Q x relT (B per head, shared over b)
    gemm(qbf, relt, rell, LL, NREL, DQ, DQ, NREL, NREL,
         NB * HH, (long)LL * DQ, (long)DQ * NREL, (long)LL * NREL, HH, 0);
    // combine + softmax -> bf16 probs
    {
      dim3 g(LL, NB * HH);
      k_softmax_rel<<<g, 256, 0, stream>>>(logit, rell, probs, inv_sqrt);
    }
    // O = probs x V
    gemm(probs, vbf, attno, LL, DQ, LL, LL, DQ, DQ,
         NB * HH, (long)LL * LL, (long)LL * DQ, (long)LL * DQ, 0, 0);
    // merge heads -> [B*L, DD] bf16
    pack(attno, obf, LL, DQ, NB * HH, HH,
         (long)HH * LL * DQ, (long)LL * DQ, DQ, 1,
         (long)LL * DD, DQ, DD, 1);
    // output projection
    gemm(obf, wot, attnO, (int)M4, DD, DD, DD, DD, DD, 1, 0, 0, 0, 0, 0);
    // x = LN(x + attn)
    k_ln<<<(unsigned)M4, 256, 0, stream>>>(xseq, attnO, ln1_g + (long)l * DD, ln1_b + (long)l * DD, xbf);

    // FFN
    gemm(xbf, wff1, ffh, (int)M4, FFD, DD, DD, FFD, FFD, 1, 0, 0, 0, 0, 0);
    {
      long tot = M4 * FFD;
      k_bias_act<<<(unsigned)((tot + 255) / 256), 256, 0, stream>>>(ffh, ff1_b + (long)l * FFD, tot, FFD, 1, ffhbf);
    }
    gemm(ffhbf, wff2, ffo, (int)M4, DD, FFD, FFD, DD, DD, 1, 0, 0, 0, 0, 0);
    {
      long tot = M4 * DD;
      k_bias_act<<<(unsigned)((tot + 255) / 256), 256, 0, stream>>>(ffo, ff2_b + (long)l * DD, tot, DD, 0, nullptr);
    }
    k_ln<<<(unsigned)M4, 256, 0, stream>>>(xseq, ffo, ln2_g + (long)l * DD, ln2_b + (long)l * DD, xbf);
  }

  // ---- output projection ----
  pack(w_out_w, woutT, NOUT, DD, 1, 1, 0, 0, DD, 1, 0, 0, 1, NOUT);
  gemm(xbf, woutT, out, (int)M4, NOUT, DD, DD, NOUT, NOUT, 1, 0, 0, 0, 0, 0);
  {
    long tot = M4 * NOUT;
    k_bias_act<<<(unsigned)((tot + 255) / 256), 256, 0, stream>>>(out, w_out_b, tot, NOUT, 0, nullptr);
  }
}
